// PlainDecoder_57466662421068
// MI455X (gfx1250) — matrix operand order repaired and checked by reading, not compiled, not run on hardware
//
#include <hip/hip_runtime.h>
#include <math.h>

// ---------------- problem constants ----------------
constexpr int NV = 32000;   // vocab
constexpr int NE = 512;     // embed dim
constexpr int NH = 512;     // hidden
constexpr int NB = 32;      // batch
constexpr int NS = 128;     // seq len
constexpr int NROWS = NS * NB;        // 4096 time-major rows
constexpr int G3 = 3 * NH;            // 1536 gate columns

// ---------------- WMMA types ----------------
typedef __bf16 bf16;
typedef __attribute__((ext_vector_type(16))) __bf16 v16bf;
typedef __attribute__((ext_vector_type(8)))  __bf16 v8bf;
typedef __attribute__((ext_vector_type(8)))  float  v8f;

__device__ __forceinline__ unsigned short f2bf(float f) {
    unsigned u = __float_as_uint(f);
    unsigned r = u + 0x7fffu + ((u >> 16) & 1u);   // round-to-nearest-even
    return (unsigned short)(r >> 16);
}

__device__ __forceinline__ v16bf pack_a(const bf16* arow, int k0, int koA) {
    // 16-bit A 16x32 layout: lane half ko in {0,8}; elems[0..7] = K=k0+2*ko..+7,
    // elems[8..15] = K=k0+2*ko+8..+7 (contiguous K per lane half, matches B)
    v8bf a0 = *(const v8bf*)(arow + k0 + 2 * koA);
    v8bf a1 = *(const v8bf*)(arow + k0 + 2 * koA + 8);
    v16bf a;
#pragma unroll
    for (int i = 0; i < 8; ++i) { a[i] = a0[i]; a[8 + i] = a1[i]; }
    return a;
}

// ---------------- conversion / gather kernels ----------------
__global__ void cvt_f32_bf16(const float* __restrict__ src,
                             unsigned short* __restrict__ dst, size_t n) {
    size_t i = (size_t)blockIdx.x * blockDim.x + threadIdx.x;
    size_t stride = (size_t)gridDim.x * blockDim.x;
    for (; i < n; i += stride) dst[i] = f2bf(src[i]);
}

// emb_tm[(s*NB+b)*NE + e] = bf16(table[target[b*NS+s]*NE + e])
__global__ void embed_bf(const int* __restrict__ tgt,
                         const float* __restrict__ tbl,
                         unsigned short* __restrict__ out) {
    int i = blockIdx.x * blockDim.x + threadIdx.x;       // < NS*NB*NE
    int e  = i & (NE - 1);
    int rb = i >> 9;                                     // NE = 512
    int b  = rb & (NB - 1);
    int s  = rb >> 5;                                    // NB = 32
    int tok = tgt[b * NS + s];
    out[i] = f2bf(tbl[(size_t)tok * NE + e]);
}

__global__ void init_h(const float* __restrict__ ctx, float* __restrict__ hf,
                       unsigned short* __restrict__ hb, int n) {
    int i = blockIdx.x * blockDim.x + threadIdx.x;
    if (i < n) { float v = ctx[i]; hf[i] = v; hb[i] = f2bf(v); }
}

// ---------------- generic GEMM: C[z] = A * W[z]^T + bias[z] ----------------
// A: [M,K] bf16 (act, row-major), W: [Z,N,K] bf16, bias: [Z,N] f32, C: [Z,M,N] f32
// grid = (M/16, Ntiles/8, Z), block = 256 (8 waves, one 16x16 tile each)
__global__ void __launch_bounds__(256)
gemm_bf16(const unsigned short* __restrict__ A, const unsigned short* __restrict__ W,
          const float* __restrict__ bias, float* __restrict__ C,
          int M, int N, int K) {
    int wave = threadIdx.x >> 5, lane = threadIdx.x & 31;
    int mt = blockIdx.x, nt = blockIdx.y * 8 + wave, z = blockIdx.z;
    int m0 = mt << 4, n0 = nt << 4;
    int cl = lane & 15, hf = lane >> 4;
    int koA = hf * 8, kbB = hf * 16;

    const bf16* Ab = (const bf16*)A + (size_t)(m0 + cl) * K;
    const bf16* Wb = (const bf16*)W + (size_t)z * N * K + (size_t)(n0 + cl) * K;
    v8f acc = {};
    for (int k0 = 0; k0 < K; k0 += 32) {
        v16bf a = pack_a(Ab, k0, koA);
        v16bf b = *(const v16bf*)(Wb + k0 + kbB);   // B 32x16: lane col, contiguous K
        acc = __builtin_amdgcn_wmma_f32_16x16x32_bf16(false, a, false, b,
                                                      (short)0, acc, false, false);
    }
    float bc = bias[(size_t)z * N + n0 + cl];
    float* Cz = C + (size_t)z * M * N;
#pragma unroll
    for (int j = 0; j < 8; ++j) {
        int row = m0 + j + hf * 8;                  // C/D: VGPR j -> rows j / j+8
        Cz[(size_t)row * N + n0 + cl] = acc[j] + bc;
    }
}

// ---------------- FC GEMM, LDS-staged + register-blocked -------------------
// Block computes a 128x128 output tile: 8 waves, each 4 M-tiles x 2 N-tiles.
// A panel (128 x 128-K-slab) staged in LDS with row padding (+8 elems => 272B
// row stride => 4-bank rotation per row => conflict-free fragment reads).
// L2 traffic: B = 32 x 65.5MB ~ 2.1GB, A = 250 x 8MB ~ 2GB (vs 33GB unblocked).
// A: l1out bf16 [4096,1024] time-major; W: fc_w bf16 [NV,1024]; out: [NB,NS,NV]
constexpr int FC_KS  = 128;          // K slab
constexpr int FC_LDA = FC_KS + 8;    // padded LDS row, in bf16 elems

__global__ void __launch_bounds__(256, 1)
fc_gemm(const unsigned short* __restrict__ A, const unsigned short* __restrict__ W,
        const float* __restrict__ bias, float* __restrict__ out) {
    constexpr int K = 2 * NH;                         // 1024
    __shared__ __align__(16) unsigned short sA[128 * FC_LDA];

    int tid  = threadIdx.x;
    int wave = tid >> 5, lane = tid & 31;
    int cl = lane & 15, hf = lane >> 4;
    int koA = hf * 8, kbB = hf * 16;

    int m00 = blockIdx.x * 128;                       // 8 m-tiles per block
    int n00 = blockIdx.y * 128;                       // 8 n-tiles per block
    int wm  = wave >> 2;                              // 0..1 -> 64 rows each
    int wn  = wave & 3;                               // 0..3 -> 32 cols each
    int nt0 = n00 + wn * 32;

    const bf16* Wb0 = (const bf16*)W + (size_t)(nt0 + cl) * K;
    const bf16* Wb1 = (const bf16*)W + (size_t)(nt0 + 16 + cl) * K;

    v8f acc[4][2] = {};
    for (int k0 = 0; k0 < K; k0 += FC_KS) {
        __syncthreads();                              // previous slab consumed
        // cooperative stage of A slab [128 x FC_KS]: 8 x 16B per thread, coalesced
#pragma unroll
        for (int i = 0; i < 8; ++i) {
            int c   = i * 256 + tid;                  // 0..2047 16B-chunks
            int row = c >> 4;
            int kc8 = (c & 15) * 8;
            *(v8bf*)((bf16*)sA + row * FC_LDA + kc8) =
                *(const v8bf*)((const bf16*)A + (size_t)(m00 + row) * K + k0 + kc8);
        }
        __syncthreads();
#pragma unroll
        for (int kc = 0; kc < FC_KS; kc += 32) {
            v16bf b0 = *(const v16bf*)(Wb0 + k0 + kc + kbB);
            v16bf b1 = *(const v16bf*)(Wb1 + k0 + kc + kbB);
#pragma unroll
            for (int i = 0; i < 4; ++i) {
                const bf16* arow = (const bf16*)sA +
                                   (size_t)(wm * 64 + i * 16 + cl) * FC_LDA + kc;
                v16bf a = pack_a(arow, 0, koA);
                acc[i][0] = __builtin_amdgcn_wmma_f32_16x16x32_bf16(
                    false, a, false, b0, (short)0, acc[i][0], false, false);
                acc[i][1] = __builtin_amdgcn_wmma_f32_16x16x32_bf16(
                    false, a, false, b1, (short)0, acc[i][1], false, false);
            }
        }
    }
    float bc0 = bias[nt0 + cl], bc1 = bias[nt0 + 16 + cl];
#pragma unroll
    for (int i = 0; i < 4; ++i) {
#pragma unroll
        for (int j = 0; j < 8; ++j) {
            int r = m00 + wm * 64 + i * 16 + j + hf * 8;  // time-major r = s*NB+b
            int b = r & (NB - 1), s = r >> 5;
            float* o = out + ((size_t)b * NS + s) * NV;
            o[nt0 + cl]      = acc[i][0][j] + bc0;
            o[nt0 + 16 + cl] = acc[i][1][j] + bc1;
        }
    }
}

// ---------------- persistent GRU recurrence (one kernel per layer) -----------
// 16 blocks x 256 thr = 128 waves: wave -> (dir d, row-tile mt, 16-col group cg).
// Each wave computes gh tiles for all 3 gates of its 16 hidden units, then the
// gate math is lane-local (gate triples share lane+acc index). Double-buffered h,
// one software grid barrier per step.
__global__ void __launch_bounds__(256, 1)
gru_recurrence(const float* __restrict__ gi,            // [2, NS*NB, 3H] f32
               const unsigned short* __restrict__ Whh,  // [2, 3H, H] bf16
               const float* __restrict__ bhh,           // [2, 3H]
               float* __restrict__ hfb,                 // [2buf][2][NB][H] f32
               unsigned short* __restrict__ hbb,        // same, bf16
               unsigned short* __restrict__ outbf,      // [NS*NB, 2H] bf16
               unsigned* __restrict__ barrier) {
    int wave = blockIdx.x * 8 + (threadIdx.x >> 5);     // 0..127
    int lane = threadIdx.x & 31;
    int d  = wave >> 6;
    int mt = (wave >> 5) & 1;
    int cg = wave & 31;
    int m0 = mt * 16, j0 = cg * 16;
    int cl = lane & 15, hf = lane >> 4;
    int koA = hf * 8, kbB = hf * 16;

    const bf16* Wd = (const bf16*)Whh + (size_t)d * G3 * NH;
    const bf16* wr = Wd + (size_t)(j0 + cl) * NH;
    const bf16* wz = Wd + (size_t)(NH + j0 + cl) * NH;
    const bf16* wn = Wd + (size_t)(2 * NH + j0 + cl) * NH;
    float bhr = bhh[d * G3 + j0 + cl];
    float bhz = bhh[d * G3 + NH + j0 + cl];
    float bhn = bhh[d * G3 + 2 * NH + j0 + cl];
    const size_t HBUF = (size_t)2 * NB * NH;            // elems per buffer
    int c = j0 + cl;

    for (int t = 0; t < NS; ++t) {
        int cur = t & 1, nxt = cur ^ 1;
        const bf16* arow = (const bf16*)hbb + (size_t)cur * HBUF +
                           (size_t)d * NB * NH + (size_t)(m0 + cl) * NH;
        v8f ar = {}, az = {}, an = {};
        for (int k0 = 0; k0 < NH; k0 += 32) {
            v16bf a  = pack_a(arow, k0, koA);
            v16bf br = *(const v16bf*)(wr + k0 + kbB);
            v16bf bz = *(const v16bf*)(wz + k0 + kbB);
            v16bf bn = *(const v16bf*)(wn + k0 + kbB);
            ar = __builtin_amdgcn_wmma_f32_16x16x32_bf16(false, a, false, br, (short)0, ar, false, false);
            az = __builtin_amdgcn_wmma_f32_16x16x32_bf16(false, a, false, bz, (short)0, az, false, false);
            an = __builtin_amdgcn_wmma_f32_16x16x32_bf16(false, a, false, bn, (short)0, an, false, false);
        }
        const float* hfc = hfb + (size_t)cur * HBUF + (size_t)d * NB * NH;
        float* hfn = hfb + (size_t)nxt * HBUF + (size_t)d * NB * NH;
        unsigned short* hbn = hbb + (size_t)nxt * HBUF + (size_t)d * NB * NH;
#pragma unroll
        for (int j = 0; j < 8; ++j) {
            int b = m0 + j + hf * 8;
            const float* gib = gi + ((size_t)d * NROWS + (size_t)t * NB + b) * G3;
            float r = 1.f / (1.f + expf(-(gib[c] + ar[j] + bhr)));
            float z = 1.f / (1.f + expf(-(gib[NH + c] + az[j] + bhz)));
            float n = tanhf(gib[2 * NH + c] + r * (an[j] + bhn));
            float hp = hfc[(size_t)b * NH + c];
            float hnew = (1.f - z) * n + z * hp;
            hfn[(size_t)b * NH + c] = hnew;
            unsigned short hb16 = f2bf(hnew);
            hbn[(size_t)b * NH + c] = hb16;
            outbf[((size_t)t * NB + b) * (2 * NH) + d * NH + c] = hb16;
        }
        // ---- software grid barrier (16 co-resident blocks) ----
        __syncthreads();
        if (threadIdx.x == 0) {
            __threadfence();
            atomicAdd(barrier, 1u);
            unsigned target = (unsigned)gridDim.x * (unsigned)(t + 1);
            while (atomicAdd(barrier, 0u) < target) { __builtin_amdgcn_s_sleep(1); }
        }
        __syncthreads();
        __threadfence();
    }
}

// ---------------- log-softmax: per-row logsumexp then subtract ---------------
__global__ void row_lse(const float* __restrict__ logits, float* __restrict__ lse) {
    __shared__ float sm[256];
    const float* p = logits + (size_t)blockIdx.x * NV;
    float m = -INFINITY;
    for (int i = threadIdx.x; i < NV; i += 256) m = fmaxf(m, p[i]);
    sm[threadIdx.x] = m; __syncthreads();
    for (int s = 128; s > 0; s >>= 1) {
        if (threadIdx.x < s) sm[threadIdx.x] = fmaxf(sm[threadIdx.x], sm[threadIdx.x + s]);
        __syncthreads();
    }
    m = sm[0]; __syncthreads();
    float sum = 0.f;
    for (int i = threadIdx.x; i < NV; i += 256) sum += expf(p[i] - m);
    sm[threadIdx.x] = sum; __syncthreads();
    for (int s = 128; s > 0; s >>= 1) {
        if (threadIdx.x < s) sm[threadIdx.x] += sm[threadIdx.x + s];
        __syncthreads();
    }
    if (threadIdx.x == 0) lse[blockIdx.x] = m + logf(sm[0]);
}

__global__ void sub_lse(float* __restrict__ logits, const float* __restrict__ lse,
                        size_t n4) {
    size_t i = (size_t)blockIdx.x * blockDim.x + threadIdx.x;
    if (i >= n4) return;
    size_t row = i / (NV / 4);
    float4 v = ((float4*)logits)[i];
    float l = lse[row];
    v.x -= l; v.y -= l; v.z -= l; v.w -= l;
    ((float4*)logits)[i] = v;
}

// ---------------- workspace layout (all 256B aligned) ----------------
constexpr size_t OFF_CNT  = 0;                                    // 2 counters
constexpr size_t OFF_EMB  = 256;                                  // 4,194,304
constexpr size_t OFF_WIH0 = OFF_EMB  + (size_t)NROWS * NE * 2;
constexpr size_t OFF_WHH0 = OFF_WIH0 + (size_t)2 * G3 * NE * 2;   // 3,145,728
constexpr size_t OFF_WIH1 = OFF_WHH0 + (size_t)2 * G3 * NH * 2;
constexpr size_t OFF_WHH1 = OFF_WIH1 + (size_t)2 * G3 * 2 * NH * 2;
constexpr size_t OFF_FCW  = OFF_WHH1 + (size_t)2 * G3 * NH * 2;
constexpr size_t OFF_GI   = OFF_FCW  + (size_t)NV * 2 * NH * 2;   // 65,536,000
constexpr size_t OFF_L0   = OFF_GI   + (size_t)2 * NROWS * G3 * 4;// 50,331,648
constexpr size_t OFF_L1   = OFF_L0   + (size_t)NROWS * 2 * NH * 2;
constexpr size_t OFF_HF   = OFF_L1   + (size_t)NROWS * 2 * NH * 2;
constexpr size_t OFF_HB   = OFF_HF   + (size_t)2 * 2 * NB * NH * 4;
constexpr size_t OFF_LSE  = OFF_HB   + (size_t)2 * 2 * NB * NH * 2;

extern "C" void kernel_launch(void* const* d_in, const int* in_sizes, int n_in,
                              void* d_out, int out_size, void* d_ws, size_t ws_size,
                              hipStream_t stream) {
    const int*   target  = (const int*)d_in[0];
    const float* context = (const float*)d_in[2];
    const float* table   = (const float*)d_in[3];
    const float* w_ih0   = (const float*)d_in[4];
    const float* w_hh0   = (const float*)d_in[5];
    const float* b_ih0   = (const float*)d_in[6];
    const float* b_hh0   = (const float*)d_in[7];
    const float* w_ih1   = (const float*)d_in[8];
    const float* w_hh1   = (const float*)d_in[9];
    const float* b_ih1   = (const float*)d_in[10];
    const float* b_hh1   = (const float*)d_in[11];
    const float* fc_w    = (const float*)d_in[12];
    const float* fc_b    = (const float*)d_in[13];
    float* out = (float*)d_out;
    char*  ws  = (char*)d_ws;

    unsigned*       cnt   = (unsigned*)(ws + OFF_CNT);
    unsigned short* emb   = (unsigned short*)(ws + OFF_EMB);
    unsigned short* wih0b = (unsigned short*)(ws + OFF_WIH0);
    unsigned short* whh0b = (unsigned short*)(ws + OFF_WHH0);
    unsigned short* wih1b = (unsigned short*)(ws + OFF_WIH1);
    unsigned short* whh1b = (unsigned short*)(ws + OFF_WHH1);
    unsigned short* fcwb  = (unsigned short*)(ws + OFF_FCW);
    float*          gi    = (float*)(ws + OFF_GI);
    unsigned short* l0out = (unsigned short*)(ws + OFF_L0);
    unsigned short* l1out = (unsigned short*)(ws + OFF_L1);
    float*          hfb   = (float*)(ws + OFF_HF);
    unsigned short* hbb   = (unsigned short*)(ws + OFF_HB);
    float*          lse   = (float*)(ws + OFF_LSE);

    hipMemsetAsync(cnt, 0, 256, stream);   // reset both grid-barrier counters

    // bf16 weight copies (one-time-per-call; HBM/L2 streaming, ~10us total)
    cvt_f32_bf16<<<4096, 256, 0, stream>>>(w_ih0, wih0b, (size_t)2 * G3 * NE);
    cvt_f32_bf16<<<4096, 256, 0, stream>>>(w_hh0, whh0b, (size_t)2 * G3 * NH);
    cvt_f32_bf16<<<4096, 256, 0, stream>>>(w_ih1, wih1b, (size_t)2 * G3 * 2 * NH);
    cvt_f32_bf16<<<4096, 256, 0, stream>>>(w_hh1, whh1b, (size_t)2 * G3 * NH);
    cvt_f32_bf16<<<8192, 256, 0, stream>>>(fc_w,  fcwb,  (size_t)NV * 2 * NH);

    embed_bf<<<(NS * NB * NE) / 256, 256, 0, stream>>>(target, table, emb);

    // Layer 0: gi0 for all steps (parallel GEMM), then persistent recurrence
    gemm_bf16<<<dim3(NROWS / 16, (G3 / 16) / 8, 2), 256, 0, stream>>>(
        emb, wih0b, b_ih0, gi, NROWS, G3, NE);
    init_h<<<128, 256, 0, stream>>>(context, hfb, hbb, 2 * NB * NH);
    gru_recurrence<<<16, 256, 0, stream>>>(gi, whh0b, b_hh0, hfb, hbb, l0out, cnt + 0);

    // Layer 1
    gemm_bf16<<<dim3(NROWS / 16, (G3 / 16) / 8, 2), 256, 0, stream>>>(
        l0out, wih1b, b_ih1, gi, NROWS, G3, 2 * NH);
    init_h<<<128, 256, 0, stream>>>(context + 2 * NB * NH, hfb, hbb, 2 * NB * NH);
    gru_recurrence<<<16, 256, 0, stream>>>(gi, whh1b, b_hh1, hfb, hbb, l1out, cnt + 1);

    // FC + log-softmax: block tile 128x128, grid (4096/128, 32000/128)
    fc_gemm<<<dim3(NROWS / 128, NV / 128), 256, 0, stream>>>(l1out, fcwb, fc_b, out);
    row_lse<<<NROWS, 256, 0, stream>>>(out, lse);
    size_t n4 = (size_t)NROWS * (NV / 4);
    sub_lse<<<(unsigned)((n4 + 255) / 256), 256, 0, stream>>>(out, lse, n4);
}
